// Net_39591008534867
// MI455X (gfx1250) — matrix operand order repaired and checked by reading, not compiled, not run on hardware
//
#include <hip/hip_runtime.h>
#include <math.h>

// MI455X / gfx1250 fused kernel for the 3-block attention + MLP network.
// One wave (32 lanes) owns 4 batches = 16 rows (4 slots/batch, slot 3 = pad).
// Block = 64 threads = 2 waves = 8 batches. Grid = B/8.
// All weights staged once per block into LDS as padded f16; every GEMM runs on
// v_wmma_f32_16x16x32_f16 with f32 accumulation.

typedef _Float16 v16h __attribute__((ext_vector_type(16)));
typedef _Float16 h8   __attribute__((ext_vector_type(8)));
typedef float    v8f  __attribute__((ext_vector_type(8)));

#define NBLK  3
#define SSD   26
#define NAD   19
#define FD    45     // SS + NA
#define VDD   128
#define OUTD  79     // NB*SS + 1

__device__ __forceinline__ v8f wmma16(v16h a, v16h b, v8f c) {
  // v_wmma_f32_16x16x32_f16  (neg_a, A, neg_b, B, c_mod, C, reuse_a, reuse_b)
  return __builtin_amdgcn_wmma_f32_16x16x32_f16(false, a, false, b, (short)0, c,
                                                false, false);
}

// A fragment (16x32 f16) from row-major LDS [16][stride].
// lane l: m = l&15 ; koff = (l>>4)*8 ; halfs 0..7 -> K=koff+h ; 8..15 -> 16+koff+h
// Both chunks are 8 contiguous halfs, 16B aligned -> ds_load_b128.
__device__ __forceinline__ v16h load_a_lds(const _Float16* base, int stride,
                                           int lane, int kc) {
  const int m    = lane & 15;
  const int koff = (lane >> 4) * 8;
  const _Float16* p = base + m * stride + kc * 32 + koff;
  const h8 lo = *(const h8*)(p);
  const h8 hi = *(const h8*)(p + 16);
  v16h a;
#pragma unroll
  for (int h = 0; h < 8; ++h) { a[h] = lo[h]; a[8 + h] = hi[h]; }
  return a;
}

// B fragment (32x16 f16) for B = W^T from padded row-major LDS weights
// [d][stride]. lane l: n = l&15 (row d = nt*16+n); koff = (l>>4)*8 ;
// halfs 0..7 -> K=koff+h ; 8..15 -> 16+koff+h (16B aligned -> 2x ds_load_b128).
__device__ __forceinline__ v16h load_b_lds(const _Float16* base, int stride,
                                           int lane, int nt, int kc) {
  const int d  = nt * 16 + (lane & 15);
  const int fb = kc * 32 + ((lane >> 4) * 8);
  const _Float16* p = base + d * stride + fb;
  const h8 lo = *(const h8*)(p);
  const h8 hi = *(const h8*)(p + 16);
  v16h b;
#pragma unroll
  for (int h = 0; h < 8; ++h) { b[h] = lo[h]; b[8 + h] = hi[h]; }
  return b;
}

// D fragment (16x16 f32): lane l reg r -> (m = r + 8*(l>>4), n = l&15)
__device__ __forceinline__ void store_d_f32(float* base, int stride, int nt,
                                            int lane, v8f c) {
  const int n  = lane & 15;
  const int mo = (lane >> 4) * 8;
#pragma unroll
  for (int r = 0; r < 8; ++r) base[(mo + r) * stride + nt * 16 + n] = c[r];
}

__device__ __forceinline__ void store_d_f16(_Float16* base, int stride, int nt,
                                            int lane, v8f c) {
  const int n  = lane & 15;
  const int mo = (lane >> 4) * 8;
#pragma unroll
  for (int r = 0; r < 8; ++r)
    base[(mo + r) * stride + nt * 16 + n] = (_Float16)c[r];
}

__global__ __launch_bounds__(64) void fused_block_attn_kernel(
    const float* __restrict__ states,    // [B,3,26]
    const float* __restrict__ block_id,  // [B,3,1]
    const float* __restrict__ action,    // [B,19]
    const float* __restrict__ Wq, const float* __restrict__ bq,   // [128,45],[128]
    const float* __restrict__ Wk, const float* __restrict__ bk,
    const float* __restrict__ Wv, const float* __restrict__ bv,
    const float* __restrict__ W1, const float* __restrict__ b1,   // [128,128],[128]
    const float* __restrict__ W2, const float* __restrict__ b2,   // [26,128],[26]
    const float* __restrict__ Wc, const float* __restrict__ bc,   // [135],[1]
    float* __restrict__ out)             // [B,79]
{
  // ---- LDS: weights (per block) + per-wave scratch. Total ~148.5 KB. ---------
  __shared__ alignas(16) _Float16 sWqkv[3][128 * 64];  // K padded 45->64 (48 KB)
  __shared__ alignas(16) _Float16 sW1[128 * 128];      // (32 KB)
  __shared__ alignas(16) _Float16 sW2[32 * 128];       // d padded 26->32 (8 KB)
  __shared__ alignas(16) _Float16 sBA[2][16 * 64];     // block_action tiles
  __shared__ alignas(16) float    sQ[2][16 * VDD];
  __shared__ alignas(16) float    sK[2][16 * VDD];
  __shared__ alignas(16) float    sV[2][16 * VDD];
  __shared__ alignas(16) _Float16 sRT[2][16 * VDD];    // resid, then tanh(h1)
  __shared__ float sAttn[2][64];  // [0..11]=|q|, [12..23]=|k|, [24..59]=a

  const int tid    = threadIdx.x;
  const int w      = tid >> 5;
  const int lane   = tid & 31;
  const int batch0 = blockIdx.x * 8 + w * 4;   // 4 batches per wave

  _Float16* ba = sBA[w];
  float *q32 = sQ[w], *k32 = sK[w], *v32 = sV[w];
  _Float16* rt16 = sRT[w];
  float* attn = sAttn[w];

  // ---- Phase 0: stage weights into LDS (f16, zero-padded) --------------------
  for (int idx = tid; idx < 128 * 64; idx += 64) {
    const int d = idx >> 6, f = idx & 63;
    float vq = 0.0f, vk = 0.0f, vv = 0.0f;
    if (f < FD) {
      const int g = d * FD + f;
      vq = Wq[g]; vk = Wk[g]; vv = Wv[g];
    }
    sWqkv[0][idx] = (_Float16)vq;
    sWqkv[1][idx] = (_Float16)vk;
    sWqkv[2][idx] = (_Float16)vv;
  }
  for (int idx = tid; idx < 128 * 128; idx += 64)
    sW1[idx] = (_Float16)W1[idx];
  for (int idx = tid; idx < 32 * 128; idx += 64) {
    const int d = idx >> 7;
    sW2[idx] = (d < SSD) ? (_Float16)W2[idx] : (_Float16)0.0f;
  }

  // ---- Phase 1: build block_action tile (16 rows x 64 cols, f16) -------------
  for (int idx = lane; idx < 16 * 64; idx += 32) {
    const int m = idx >> 6, col = idx & 63;
    const int b = batch0 + (m >> 2), slot = m & 3;
    float v = 0.0f;
    if (slot < 3) {
      if (col < SSD) {
        v = states[(b * NBLK + slot) * SSD + col];
      } else if (col < FD) {
        const float blk = block_id[b * NBLK + slot];
        v = (blk == 1.0f) ? action[b * NAD + (col - SSD)] : -1.0f;
      }
    }
    ba[m * 64 + col] = (_Float16)v;
  }
  __syncthreads();

  // ---- Phase 2: QKV projections (16x45 x 45x128, K padded to 64) -------------
  {
    const v16h a0 = load_a_lds(ba, 64, lane, 0);
    const v16h a1 = load_a_lds(ba, 64, lane, 1);
    const float* Bt[3] = {bq, bk, bv};
    float*       Ot[3] = {q32, k32, v32};
#pragma unroll
    for (int nt = 0; nt < 8; ++nt) {
      const int d = nt * 16 + (lane & 15);
#pragma unroll
      for (int t = 0; t < 3; ++t) {
        const float bias = Bt[t][d];
        v8f c;
#pragma unroll
        for (int r = 0; r < 8; ++r) c[r] = bias;
        c = wmma16(a0, load_b_lds(sWqkv[t], 64, lane, nt, 0), c);
        c = wmma16(a1, load_b_lds(sWqkv[t], 64, lane, nt, 1), c);
        store_d_f32(Ot[t], VDD, nt, lane, c);
      }
    }
  }
  __syncthreads();

  // ---- Phase 3a: q/k norms (24 lanes, one row each) --------------------------
  if (lane < 24) {
    const int isK = (lane >= 12) ? 1 : 0;
    const int idx = lane - isK * 12;
    const int m   = (idx / 3) * 4 + (idx % 3);
    const float* src = isK ? k32 : q32;
    float s = 0.0f;
    for (int dd = 0; dd < VDD; ++dd) {
      const float x = src[m * VDD + dd];
      s += x * x;
    }
    attn[lane] = sqrtf(s);
  }
  __syncthreads();

  // ---- Phase 3b: cosine similarity + softmax (12 lanes, one row each) --------
  if (lane < 12) {
    const int sb = lane / 3, i = lane % 3, mi = sb * 4 + i;
    const float qn = attn[lane];
    float c3[3];
#pragma unroll
    for (int j = 0; j < 3; ++j) {
      const int mj = sb * 4 + j;
      float dot = 0.0f;
      for (int dd = 0; dd < VDD; ++dd)
        dot += q32[mi * VDD + dd] * k32[mj * VDD + dd];
      const float kn = attn[12 + sb * 3 + j];
      c3[j] = dot / fmaxf(qn * kn, 1e-8f);
    }
    const float mx = fmaxf(c3[0], fmaxf(c3[1], c3[2]));
    const float e0 = expf(c3[0] - mx), e1 = expf(c3[1] - mx), e2 = expf(c3[2] - mx);
    const float inv = 1.0f / (e0 + e1 + e2);
    attn[24 + lane * 3 + 0] = e0 * inv;
    attn[24 + lane * 3 + 1] = e1 * inv;
    attn[24 + lane * 3 + 2] = e2 * inv;
  }
  __syncthreads();

  // ---- Phase 3c: residual = a @ v  -> f16 tile -------------------------------
  for (int idx = lane; idx < 16 * VDD; idx += 32) {
    const int m = idx >> 7, dd = idx & 127;
    const int slot = m & 3, sb = m >> 2;
    float v = 0.0f;
    if (slot < 3) {
      const int rr = sb * 3 + slot;
      const float a0 = attn[24 + rr * 3 + 0];
      const float a1 = attn[24 + rr * 3 + 1];
      const float a2 = attn[24 + rr * 3 + 2];
      v = a0 * v32[(sb * 4 + 0) * VDD + dd] +
          a1 * v32[(sb * 4 + 1) * VDD + dd] +
          a2 * v32[(sb * 4 + 2) * VDD + dd];
    }
    rt16[m * VDD + dd] = (_Float16)v;
  }
  __syncthreads();

  // ---- Phase 4: h1 = tanh(resid @ W1^T + b1)  (16x128x128) -------------------
  // A-fragments are fully loaded into registers BEFORE the first store, so the
  // tanh output can safely overwrite the residual buffer (LDS ops are in-order
  // within a wave).
  {
    v16h ra[4];
#pragma unroll
    for (int kc = 0; kc < 4; ++kc) ra[kc] = load_a_lds(rt16, VDD, lane, kc);
#pragma unroll
    for (int nt = 0; nt < 8; ++nt) {
      const int d = nt * 16 + (lane & 15);
      const float bias = b1[d];
      v8f c;
#pragma unroll
      for (int r = 0; r < 8; ++r) c[r] = bias;
#pragma unroll
      for (int kc = 0; kc < 4; ++kc)
        c = wmma16(ra[kc], load_b_lds(sW1, VDD, lane, nt, kc), c);
#pragma unroll
      for (int r = 0; r < 8; ++r) c[r] = tanhf(c[r]);
      store_d_f16(rt16, VDD, nt, lane, c);
    }
  }
  __syncthreads();

  // ---- Phase 5: h2 = h1 @ W2^T + b2, then out = states + h2 ------------------
  {
    v16h ta[4];
#pragma unroll
    for (int kc = 0; kc < 4; ++kc) ta[kc] = load_a_lds(rt16, VDD, lane, kc);
#pragma unroll
    for (int nt = 0; nt < 2; ++nt) {
      const int d = nt * 16 + (lane & 15);
      const bool dv = (d < SSD);
      const float bias = dv ? b2[d] : 0.0f;
      v8f c;
#pragma unroll
      for (int r = 0; r < 8; ++r) c[r] = bias;
#pragma unroll
      for (int kc = 0; kc < 4; ++kc)
        c = wmma16(ta[kc], load_b_lds(sW2, VDD, lane, nt, kc), c);
      if (dv) {
#pragma unroll
        for (int r = 0; r < 8; ++r) {
          const int m = r + (lane >> 4) * 8;
          const int slot = m & 3;
          if (slot < 3) {
            const int b = batch0 + (m >> 2);
            out[b * OUTD + slot * SSD + d] =
                c[r] + states[(b * NBLK + slot) * SSD + d];
          }
        }
      }
    }
  }

  // ---- Phase 6: confidence = sigmoid(ba_flat . Wc + bc) -> out[...,78] -------
  if (lane < 4) {
    const int b = batch0 + lane;
    float s = bc[0];
    int p = 0;
#pragma unroll
    for (int n = 0; n < NBLK; ++n) {
      const float* st = states + (b * NBLK + n) * SSD;
      for (int c2 = 0; c2 < SSD; ++c2) s += st[c2] * Wc[p++];
      const float blk = block_id[b * NBLK + n];
      const float* ac = action + b * NAD;
      for (int c2 = 0; c2 < NAD; ++c2) {
        const float v = (blk == 1.0f) ? ac[c2] : -1.0f;
        s += v * Wc[p++];
      }
    }
    out[b * OUTD + 78] = 1.0f / (1.0f + expf(-s));
  }
}

extern "C" void kernel_launch(void* const* d_in, const int* in_sizes, int n_in,
                              void* d_out, int out_size, void* d_ws, size_t ws_size,
                              hipStream_t stream) {
  (void)n_in; (void)d_ws; (void)ws_size; (void)out_size;
  const float* states   = (const float*)d_in[0];
  const float* block_id = (const float*)d_in[1];
  const float* action   = (const float*)d_in[2];
  const float* Wq = (const float*)d_in[3];  const float* bq = (const float*)d_in[4];
  const float* Wk = (const float*)d_in[5];  const float* bk = (const float*)d_in[6];
  const float* Wv = (const float*)d_in[7];  const float* bv = (const float*)d_in[8];
  const float* W1 = (const float*)d_in[9];  const float* b1 = (const float*)d_in[10];
  const float* W2 = (const float*)d_in[11]; const float* b2 = (const float*)d_in[12];
  const float* Wc = (const float*)d_in[13]; const float* bc = (const float*)d_in[14];
  float* outp = (float*)d_out;

  const int B = in_sizes[2] / NAD;     // 65536
  const int nBlocks = B / 8;           // 8 batches per 64-thread block
  fused_block_attn_kernel<<<nBlocks, 64, 0, stream>>>(
      states, block_id, action, Wq, bq, Wk, bk, Wv, bv, W1, b1, W2, b2, Wc, bc,
      outp);
}
